// Router_32968168964721
// MI455X (gfx1250) — compile-verified, matrix operand order checked by reading, not run on hardware
//
#include <hip/hip_runtime.h>
#include <hip/hip_bf16.h>

typedef __attribute__((ext_vector_type(16))) __bf16         v16bf;
typedef __attribute__((ext_vector_type(8)))  float          v8f;
typedef __attribute__((ext_vector_type(8)))  unsigned short v8us;

#define T_TOKENS  65536
#define DIM       1024
#define NEXP      256
#define MTILE     32
#define KCHUNK    64
#define NCHUNK    (DIM / KCHUNK)
#define LDS_STR   72    // 64 + 8 pad -> 144B row stride (16B aligned, bank-spread)
#define SC_STR    264   // score row stride (floats): bank = (8m + e) % 64
#define CAND_STR  65    // candidate row stride: conflict-free column scans

__device__ __forceinline__ unsigned short bf16_rne(float f) {
  unsigned int u = __float_as_uint(f);
  u += 0x7FFFu + ((u >> 16) & 1u);
  return (unsigned short)(u >> 16);
}

// ---- prepass: weight f32 -> bf16 into workspace -----------------------------
__global__ __launch_bounds__(256) void Router_cvt_w_bf16(
    const float* __restrict__ w, unsigned short* __restrict__ wb, int n) {
  int i = (blockIdx.x * 256 + threadIdx.x) * 4;
  if (i < n) {
    float4 f = *(const float4*)(w + i);
    ushort4 h;
    h.x = bf16_rne(f.x); h.y = bf16_rne(f.y);
    h.z = bf16_rne(f.z); h.w = bf16_rne(f.w);
    *(ushort4*)(wb + i) = h;
  }
}

// ---- main kernel ------------------------------------------------------------
struct Stage {
  unsigned short xs[MTILE][LDS_STR];   // bf16 x tile
  unsigned short ws[NEXP][LDS_STR];    // bf16 weight tile (async-filled)
};

struct __align__(16) LdsT {
  union {
    Stage st[2];                       // ping-pong staging
    struct {
      float sc[MTILE][SC_STR];         // sigmoid scores
      float cv[MTILE][CAND_STR];       // top-8 candidates (values)
      int   ci[MTILE][CAND_STR];       // top-8 candidates (indices)
    } tk;
  };
};

__device__ __forceinline__ void fill_stage(Stage* s,
                                           const float* __restrict__ x,
                                           const unsigned short* __restrict__ wbf,
                                           int m0, int k0, int tid, int xr, int xc) {
  // weight tile: pure bf16 copy -> async global->LDS (ASYNCcnt path, overlaps WMMA)
  const unsigned ws_lds = (unsigned)(size_t)&s->ws[tid][0];   // low 32b = LDS offset
  const unsigned short* gp = wbf + (size_t)tid * DIM + k0;
#pragma unroll
  for (int j = 0; j < 8; j++) {
    asm volatile("global_load_async_to_lds_b128 %0, %1, off"
                 :
                 : "v"(ws_lds + 16u * j),
                   "v"((unsigned long long)(size_t)(gp + 8 * j))
                 : "memory");
  }
  // x tile: f32 global -> VALU bf16 convert -> LDS
#pragma unroll
  for (int j = 0; j < 2; j++) {
    const float4 f = *(const float4*)(x + (size_t)(m0 + xr) * DIM + k0 + xc + 4 * j);
    ushort4 h;
    h.x = bf16_rne(f.x); h.y = bf16_rne(f.y);
    h.z = bf16_rne(f.z); h.w = bf16_rne(f.w);
    *(ushort4*)&s->xs[xr][xc + 4 * j] = h;
  }
}

__global__ __launch_bounds__(256, 1) void Router_main(
    const float* __restrict__ x, const unsigned short* __restrict__ wbf,
    const float* __restrict__ bias, float* __restrict__ out_w,
    int* __restrict__ out_i) {
  __shared__ LdsT lds;

  const int tid  = threadIdx.x;
  const int wave = tid >> 5;
  const int lane = tid & 31;
  const int half = lane >> 4;     // K-offset selector per ISA 16-bit layout
  const int r    = lane & 15;
  const int m0   = blockIdx.x * MTILE;

  const int m_base = (wave & 1) * 16;    // 2 m-tiles of 16 tokens
  const int n_base = (wave >> 1) * 64;   // 4 n-quarters of 64 experts

  v8f zero = {};
  v8f acc[4];
#pragma unroll
  for (int t = 0; t < 4; t++) acc[t] = zero;

  const int xr = tid >> 3;           // 0..31 token row for x staging
  const int xc = (tid & 7) * 8;      // 8 floats per thread per chunk

  // prologue: fill buffer 0 with chunk 0
  fill_stage(&lds.st[0], x, wbf, m0, 0, tid, xr, xc);
  asm volatile("s_wait_asynccnt 0" ::: "memory");
  __syncthreads();

  int c = 0;
  for (int i = 0; i < NCHUNK; i++) {
    // overlap: start filling the other buffer with chunk i+1 (WAR-safe: readers
    // of that buffer retired at the barrier ending iteration i-1)
    if (i + 1 < NCHUNK)
      fill_stage(&lds.st[c ^ 1], x, wbf, m0, (i + 1) * KCHUNK, tid, xr, xc);

    const Stage* cur = &lds.st[c];
#pragma unroll
    for (int ks = 0; ks < 2; ks++) {
      const int kb = ks * 32;
      union Frag { v16bf v; v8us h[2]; };
      Frag a;
      a.h[0] = *(const v8us*)&cur->xs[m_base + r][kb + half * 8];
      a.h[1] = *(const v8us*)&cur->xs[m_base + r][kb + 16 + half * 8];
      Frag b[4];
#pragma unroll
      for (int t = 0; t < 4; t++) {
        b[t].h[0] = *(const v8us*)&cur->ws[n_base + t * 16 + r][kb + half * 8];
        b[t].h[1] = *(const v8us*)&cur->ws[n_base + t * 16 + r][kb + 16 + half * 8];
      }
#pragma unroll
      for (int t = 0; t < 4; t++) {
        acc[t] = __builtin_amdgcn_wmma_f32_16x16x32_bf16(
            false, a.v, false, b[t].v, (short)0, acc[t], false, false);
      }
    }

    // publish next chunk's fill: drain this wave's async copies, then barrier
    asm volatile("s_wait_asynccnt 0" ::: "memory");
    __syncthreads();
    c ^= 1;
  }

  // bias + sigmoid, scatter to LDS scores (C layout: M = q + 8*half, N = r)
#pragma unroll
  for (int t = 0; t < 4; t++) {
    const int e = n_base + t * 16 + r;
    const float be = bias[e];
#pragma unroll
    for (int q = 0; q < 8; q++) {
      const int m = m_base + half * 8 + q;
      const float z = acc[t][q] + be;
      lds.tk.sc[m][e] = 1.0f / (1.0f + __expf(-z));
    }
  }
  __syncthreads();

  // ---- parallel top-8: phase 1, 8 lanes per token, stride-8 expert slices ---
  const int tm = tid >> 3;     // token within tile
  const int pp = tid & 7;      // partition
  float v[8]; int ix[8];
#pragma unroll
  for (int j = 0; j < 8; j++) { v[j] = -3.0e38f; ix[j] = 0x7fffffff; }
  for (int j = 0; j < 32; j++) {
    const int e = pp + 8 * j;
    float cv = lds.tk.sc[tm][e];
    int   ci = e;
#pragma unroll
    for (int q = 0; q < 8; q++) {
      const bool gt = (cv > v[q]) || ((cv == v[q]) && (ci < ix[q]));
      const float tv = v[q]; const int ti = ix[q];
      v[q]  = gt ? cv : v[q];
      ix[q] = gt ? ci : ix[q];
      cv = gt ? tv : cv;
      ci = gt ? ti : ci;
    }
  }
#pragma unroll
  for (int j = 0; j < 8; j++) {
    lds.tk.cv[tm][pp * 8 + j] = v[j];
    lds.tk.ci[tm][pp * 8 + j] = ix[j];
  }
  __syncthreads();

  // ---- phase 2: 32 lanes merge 64 candidates per token ----------------------
  if (tid < MTILE) {
#pragma unroll
    for (int j = 0; j < 8; j++) { v[j] = -3.0e38f; ix[j] = 0x7fffffff; }
    for (int cnd = 0; cnd < 64; cnd++) {
      float cv = lds.tk.cv[tid][cnd];
      int   ci = lds.tk.ci[tid][cnd];
#pragma unroll
      for (int q = 0; q < 8; q++) {
        const bool gt = (cv > v[q]) || ((cv == v[q]) && (ci < ix[q]));
        const float tv = v[q]; const int ti = ix[q];
        v[q]  = gt ? cv : v[q];
        ix[q] = gt ? ci : ix[q];
        cv = gt ? tv : cv;
        ci = gt ? ti : ci;
      }
    }
    float s = 0.0f;
#pragma unroll
    for (int j = 0; j < 8; j++) s += v[j];
    const float scale = 2.5f / (s + 1e-6f);
    const size_t row = (size_t)(m0 + tid);
#pragma unroll
    for (int j = 0; j < 8; j++) {
      out_w[row * 8 + j] = v[j] * scale;
      out_i[row * 8 + j] = ix[j];
    }
  }
}

extern "C" void kernel_launch(void* const* d_in, const int* in_sizes, int n_in,
                              void* d_out, int out_size, void* d_ws, size_t ws_size,
                              hipStream_t stream) {
  const float* x    = (const float*)d_in[0];
  const float* w    = (const float*)d_in[1];
  const float* bias = (const float*)d_in[2];

  unsigned short* wbf = (unsigned short*)d_ws;       // 256*1024*2 = 512 KB
  float* out_w = (float*)d_out;                      // [T,8] f32
  int*   out_i = (int*)((float*)d_out + (size_t)T_TOKENS * 8);  // [T,8] i32

  const int nw = in_sizes[1];                        // 262144
  Router_cvt_w_bf16<<<(nw / 4 + 255) / 256, 256, 0, stream>>>(w, wbf, nw);
  Router_main<<<T_TOKENS / MTILE, 256, 0, stream>>>(x, wbf, bias, out_w, out_i);
}
